// MultiHeadSlotAttention_88957362634830
// MI455X (gfx1250) — matrix operand order repaired and checked by reading, not compiled, not run on hardware
//
#include <hip/hip_runtime.h>
#include <hip/hip_bf16.h>

#define EPS_ 1e-8f
#define SCALE_ 0.125f   // DPH=64 -> 64^-0.5

typedef __attribute__((ext_vector_type(16))) __bf16 v16bf;
typedef __attribute__((ext_vector_type(8)))  float  v8f;
typedef int i32x4 __attribute__((vector_size(16)));

#if defined(__has_builtin)
#  if __has_builtin(__builtin_amdgcn_global_load_async_to_lds_b128)
#    define HAVE_ASYNC_LDS 1
#  endif
#endif
#ifndef HAVE_ASYNC_LDS
#  define HAVE_ASYNC_LDS 0
#endif

// 16-byte global->LDS copy: async (ASYNCcnt) when available, else direct.
__device__ __forceinline__ void cp16_g2l(const void* g, void* l) {
#if HAVE_ASYNC_LDS
  __builtin_amdgcn_global_load_async_to_lds_b128(
      (__attribute__((address_space(1))) i32x4*)(size_t)g,
      (__attribute__((address_space(3))) i32x4*)(unsigned)(size_t)l, 0, 0);
#else
  *(uint4*)l = *(const uint4*)g;
#endif
}

__device__ __forceinline__ void wait_async_lds() {
#if HAVE_ASYNC_LDS
#  if __has_builtin(__builtin_amdgcn_s_wait_asynccnt)
  __builtin_amdgcn_s_wait_asynccnt(0);
#  else
  asm volatile("s_wait_asynccnt 0" ::: "memory");
#  endif
#endif
}

__device__ __forceinline__ float wred32(float v) {
#pragma unroll
  for (int m = 16; m >= 1; m >>= 1) v += __shfl_xor(v, m, 32);
  return v;
}

// A matrix 16x32 (MxK) bf16 fragment: src row-major [M][K], row stride ld.
__device__ __forceinline__ v16bf load_a_frag(const __bf16* src, int ld, int kc) {
  const int lane = threadIdx.x & 31;
  const int m = lane & 15;
  const int k0 = kc + ((lane < 16) ? 0 : 16);
  const __bf16* p = src + (long)m * ld + k0;
  v16bf a;
#pragma unroll
  for (int e = 0; e < 8; ++e) a[e] = p[e];
#pragma unroll
  for (int e = 0; e < 8; ++e) a[8 + e] = p[8 + e];
  return a;
}

// B matrix 32x16 (KxN) bf16 fragment from "transposed" storage src[N][K], row stride ld.
__device__ __forceinline__ v16bf load_b_frag(const __bf16* src, int ld, int kc) {
  const int lane = threadIdx.x & 31;
  const int n = lane & 15;
  const int k0 = kc + ((lane < 16) ? 0 : 16);
  const __bf16* p = src + (long)n * ld + k0;
  v16bf b;
#pragma unroll
  for (int e = 0; e < 16; ++e) b[e] = p[e];
  return b;
}

__device__ __forceinline__ v8f wmma_bf16(v16bf a, v16bf b, v8f c) {
  return __builtin_amdgcn_wmma_f32_16x16x32_bf16(false, a, false, b, (short)0, c, false, false);
}

// ---------------- prep: transpose-convert Wq/Wk/Wv to bf16, copy slots ----------------
__global__ void k_prep(const float* Wq, const float* Wk, const float* Wv,
                       const float* slots_init,
                       __bf16* WqT, __bf16* WkT, __bf16* WvT, float* slots) {
  int idx = blockIdx.x * blockDim.x + threadIdx.x;  // 0..262143
  int m = idx & 65535;
  int sel = idx >> 16;
  if (sel < 3) {
    const float* W = (sel == 0) ? Wq : (sel == 1 ? Wk : Wv);
    __bf16* WT = (sel == 0) ? WqT : (sel == 1 ? WkT : WvT);
    WT[(m & 255) * 256 + (m >> 8)] = (__bf16)W[m];   // [out][in]
  } else {
    slots[m] = slots_init[m];
  }
}

// ---------------- LN(inputs) + K/V projection (WMMA bf16) ----------------
__global__ void k_ln_project_kv(const float* __restrict__ inputs,
                                const float* __restrict__ lng, const float* __restrict__ lnb,
                                const __bf16* __restrict__ WkT, const __bf16* __restrict__ WvT,
                                const float* __restrict__ bk, const float* __restrict__ bv,
                                __bf16* __restrict__ Kbf, __bf16* __restrict__ Vt) {
  __shared__ __bf16 xln[128 * 256];  // 64 KB
  const int tid = threadIdx.x, lane = tid & 31, wave = tid >> 5;
  const long tok0 = (long)blockIdx.x * 128;

  for (int t = wave; t < 128; t += 8) {
    const float* row = inputs + (tok0 + t) * 256 + lane * 8;
    float xv[8];
#pragma unroll
    for (int e = 0; e < 8; ++e) xv[e] = row[e];
    float s = 0.f, ss = 0.f;
#pragma unroll
    for (int e = 0; e < 8; ++e) { s += xv[e]; ss += xv[e] * xv[e]; }
    s = wred32(s); ss = wred32(ss);
    float mu = s * (1.f / 256.f);
    float var = ss * (1.f / 256.f) - mu * mu;
    float rs = rsqrtf(var + 1e-5f);
#pragma unroll
    for (int e = 0; e < 8; ++e) {
      int d = lane * 8 + e;
      xln[t * 256 + d] = (__bf16)((xv[e] - mu) * rs * lng[d] + lnb[d]);
    }
  }
  __syncthreads();

  const int b = (int)(tok0 >> 12);
  const int jbase = (int)(tok0 & 4095);
  for (int job = wave; job < 256; job += 8) {         // rt(8) x ct(16) x mat(2)
    const int mat = job & 1, ct = (job >> 1) & 15, rt = job >> 5;
    const __bf16* WT = mat ? WvT : WkT;
    v8f acc = {};
    for (int kc = 0; kc < 256; kc += 32) {
      v16bf a = load_a_frag(xln + rt * 16 * 256, 256, kc);
      v16bf bb = load_b_frag(WT + ct * 16 * 256, 256, kc);
      acc = wmma_bf16(a, bb, acc);
    }
    const int n = lane & 15;
    const int mbase = (lane < 16) ? 0 : 8;
    const int c = ct * 16 + n;
    if (mat == 0) {
      const float bias = bk[c];
#pragma unroll
      for (int r = 0; r < 8; ++r) {
        long tok = tok0 + rt * 16 + mbase + r;
        Kbf[tok * 256 + c] = (__bf16)(acc[r] + bias);
      }
    } else {
      const float bias = bv[c];
      const int h = c >> 6, dph = c & 63;
      __bf16* basep = Vt + (((long)(b * 4 + h) * 64 + dph) << 12);
#pragma unroll
      for (int r = 0; r < 8; ++r) {
        int j = jbase + rt * 16 + mbase + r;
        basep[j] = (__bf16)(acc[r] + bias);
      }
    }
  }
}

__global__ void k_zero_us(float* U, float* S) {
  int idx = blockIdx.x * blockDim.x + threadIdx.x;
  if (idx < 65536) U[idx] = 0.f;
  if (idx < 1024) S[idx] = 0.f;
}

// ---------------- LN(slots) + q projection, padded to 16 rows ----------------
__global__ void k_slots_q(const float* __restrict__ slots,
                          const float* __restrict__ lnsg, const float* __restrict__ lnsb,
                          const __bf16* __restrict__ WqT, const float* __restrict__ bq,
                          __bf16* __restrict__ qbf) {
  __shared__ __bf16 sln[16 * 256];
  const int tid = threadIdx.x, lane = tid & 31, wave = tid >> 5;
  const int b = blockIdx.x;
  for (int idx = 2048 + tid; idx < 4096; idx += 256) sln[idx] = (__bf16)0.f;  // rows 8..15 zero
  {
    const float* row = slots + ((long)b * 8 + wave) * 256 + lane * 8;
    float xv[8];
#pragma unroll
    for (int e = 0; e < 8; ++e) xv[e] = row[e];
    float s = 0.f, ss = 0.f;
#pragma unroll
    for (int e = 0; e < 8; ++e) { s += xv[e]; ss += xv[e] * xv[e]; }
    s = wred32(s); ss = wred32(ss);
    float mu = s * (1.f / 256.f), var = ss * (1.f / 256.f) - mu * mu;
    float rs = rsqrtf(var + 1e-5f);
#pragma unroll
    for (int e = 0; e < 8; ++e) {
      int d = lane * 8 + e;
      sln[wave * 256 + d] = (__bf16)((xv[e] - mu) * rs * lnsg[d] + lnsb[d]);
    }
  }
  __syncthreads();
  for (int ct = wave * 2; ct < wave * 2 + 2; ++ct) {
    v8f acc = {};
    for (int kc = 0; kc < 256; kc += 32) {
      v16bf a = load_a_frag(sln, 256, kc);
      v16bf bb = load_b_frag(WqT + ct * 16 * 256, 256, kc);
      acc = wmma_bf16(a, bb, acc);
    }
    const int n = lane & 15, mbase = (lane < 16) ? 0 : 8, c = ct * 16 + n;
    const float bias = bq[c];
#pragma unroll
    for (int r = 0; r < 8; ++r) {
      int row = mbase + r;
      float v = (row < 8) ? (acc[r] + bias) : 0.f;
      qbf[((long)b * 16 + row) * 256 + c] = (__bf16)v;
    }
  }
}

// ---------------- dots (WMMA, K staged via async LDS) + joint softmax + updates ----------
__global__ void k_attn_update(const __bf16* __restrict__ qbf, const __bf16* __restrict__ Kbf,
                              const __bf16* __restrict__ Vt,
                              float* __restrict__ U, float* __restrict__ S) {
  __shared__ __bf16 Kst[128 * 256];    // 64 KB: K chunk [128 local tokens][256 dims]
  __shared__ float dots[32 * 128];     // 16 KB, row = i*4+h
  __shared__ __bf16 attn[64 * 128];    // 16 KB, row = i*4+h for i in 0..15 (rows >=32 stay 0)
  const int tid = threadIdx.x, lane = tid & 31, wave = tid >> 5;
  const int b = blockIdx.x >> 5;
  const int j0 = (blockIdx.x & 31) * 128;

  // fire off async K-chunk copy (4096 x 16B), overlap with attn zero-init
  {
    const __bf16* gK = Kbf + ((long)b * 4096 + j0) * 256;
#pragma unroll 4
    for (int c = tid; c < 4096; c += 256) cp16_g2l(gK + c * 8, Kst + c * 8);
  }
  // prefetch the V^T chunk this block will consume in the updates stage
  {
    const int h = tid >> 6, d = tid & 63;
    const __bf16* p = Vt + (((long)(b * 4 + h) * 64 + d) << 12) + j0;
    __builtin_prefetch((const void*)p, 0, 1);
    __builtin_prefetch((const void*)(p + 64), 0, 1);
  }
  for (int idx = tid; idx < 64 * 128; idx += 256) attn[idx] = (__bf16)0.f;
  wait_async_lds();
  __syncthreads();

  for (int job = wave; job < 32; job += 8) {        // h(4) x jt(8)
    const int h = job & 3, jt = job >> 2;
    v8f acc = {};
    for (int kc = 0; kc < 64; kc += 32) {
      v16bf a = load_a_frag(qbf + (long)b * 16 * 256 + h * 64, 256, kc);
      v16bf bb = load_b_frag(Kst + (jt * 16) * 256 + h * 64, 256, kc);
      acc = wmma_bf16(a, bb, acc);
    }
    const int n = lane & 15, mbase = (lane < 16) ? 0 : 8;
#pragma unroll
    for (int r = 0; r < 8; ++r) {
      int i = mbase + r;
      if (i < 8) dots[(i * 4 + h) * 128 + jt * 16 + n] = acc[r] * SCALE_;
    }
  }
  __syncthreads();

  if (tid < 128) {                                  // softmax over the 32 (slot,head) rows
    float m = -1e30f;
#pragma unroll
    for (int r = 0; r < 32; ++r) m = fmaxf(m, dots[r * 128 + tid]);
    float p[32]; float sum = 0.f;
#pragma unroll
    for (int r = 0; r < 32; ++r) { p[r] = __expf(dots[r * 128 + tid] - m); sum += p[r]; }
    float inv = 1.f / sum;
#pragma unroll
    for (int r = 0; r < 32; ++r) {
      float v = p[r] * inv + EPS_;
      dots[r * 128 + tid] = v;
      attn[r * 128 + tid] = (__bf16)v;
    }
  }
  __syncthreads();

  if (tid < 32) {                                   // partial S per (i,h)
    float s = 0.f;
    for (int j = 0; j < 128; ++j) s += dots[tid * 128 + j];
    atomicAdd(&S[b * 32 + tid], s);
  }

  for (int job = wave; job < 16; job += 8) {        // h(4) x dt(4): U += attn @ V
    const int h = job & 3, dt = job >> 2;
    v8f acc = {};
    for (int jc = 0; jc < 128; jc += 32) {
      v16bf a = load_a_frag((const __bf16*)attn + h * 128, 512, jc);
      v16bf bb = load_b_frag(Vt + (((long)(b * 4 + h) * 64 + dt * 16) << 12) + j0, 4096, jc);
      acc = wmma_bf16(a, bb, acc);
    }
    const int n = lane & 15;
    if (lane < 16) {
#pragma unroll
      for (int r = 0; r < 8; ++r)
        atomicAdd(&U[(((long)b * 8 + r) * 4 + h) * 64 + dt * 16 + n], acc[r]);
    }
  }
}

// ---------------- updates=U/S -> GRU -> LN -> MLP residual (VALU f32, tiny) ------------
__global__ void k_gru_mlp(const float* __restrict__ U, const float* __restrict__ S,
                          float* __restrict__ slots,
                          const float* __restrict__ W_ih, const float* __restrict__ b_ih,
                          const float* __restrict__ W_hh, const float* __restrict__ b_hh,
                          const float* __restrict__ lnfg, const float* __restrict__ lnfb,
                          const float* __restrict__ W1, const float* __restrict__ b1,
                          const float* __restrict__ W2, const float* __restrict__ b2,
                          float* __restrict__ out, int write_out) {
  __shared__ float u_s[8 * 256];
  __shared__ float h_s[8 * 256];
  __shared__ float sn_s[8 * 256];
  __shared__ float lnr_s[8 * 256];
  __shared__ float hid_s[8 * 512];
  const int tid = threadIdx.x, lane = tid & 31, wave = tid >> 5;
  const int b = blockIdx.x;

  for (int idx = tid; idx < 2048; idx += 256) {
    int i = idx >> 8, dim = idx & 255, h = dim >> 6;
    float denom = S[b * 32 + i * 4 + h];
    u_s[idx] = U[(long)b * 2048 + idx] / denom;
    h_s[idx] = slots[(long)b * 2048 + idx];
  }
  __syncthreads();

  for (int p = 0; p < 8; ++p) {                     // GRU gates, 8 (i,cc) pairs/thread
    int idx = tid + p * 256;
    int i = idx >> 8, cc = idx & 255;
    const float* ur = u_s + i * 256;
    const float* hr = h_s + i * 256;
    float ir = b_ih[cc], iz = b_ih[256 + cc], in = b_ih[512 + cc];
    float hrv = b_hh[cc], hz = b_hh[256 + cc], hn = b_hh[512 + cc];
    const float* wir = W_ih + (long)cc * 256;
    const float* wiz = W_ih + (long)(256 + cc) * 256;
    const float* win = W_ih + (long)(512 + cc) * 256;
    const float* whr = W_hh + (long)cc * 256;
    const float* whz = W_hh + (long)(256 + cc) * 256;
    const float* whn = W_hh + (long)(512 + cc) * 256;
    for (int d = 0; d < 256; ++d) {
      float uv = ur[d], hv = hr[d];
      ir += uv * wir[d]; iz += uv * wiz[d]; in += uv * win[d];
      hrv += hv * whr[d]; hz += hv * whz[d]; hn += hv * whn[d];
    }
    float r = 1.f / (1.f + __expf(-(ir + hrv)));
    float z = 1.f / (1.f + __expf(-(iz + hz)));
    float ng = tanhf(in + r * hn);
    sn_s[idx] = (1.f - z) * ng + z * hr[cc];
  }
  __syncthreads();

  {                                                 // LN of 8 new-slot rows, wave per row
    const float* row = sn_s + wave * 256 + lane * 8;
    float xv[8];
#pragma unroll
    for (int e = 0; e < 8; ++e) xv[e] = row[e];
    float s = 0.f, ss = 0.f;
#pragma unroll
    for (int e = 0; e < 8; ++e) { s += xv[e]; ss += xv[e] * xv[e]; }
    s = wred32(s); ss = wred32(ss);
    float mu = s * (1.f / 256.f), var = ss * (1.f / 256.f) - mu * mu;
    float rs = rsqrtf(var + 1e-5f);
#pragma unroll
    for (int e = 0; e < 8; ++e) {
      int d = lane * 8 + e;
      lnr_s[wave * 256 + d] = (xv[e] - mu) * rs * lnfg[d] + lnfb[d];
    }
  }
  __syncthreads();

  for (int idx = tid; idx < 4096; idx += 256) {     // hidden = relu(ln @ W1 + b1)
    int i = idx >> 9, c = idx & 511;
    float acc = b1[c];
    const float* lr = lnr_s + i * 256;
    for (int d = 0; d < 256; ++d) acc += lr[d] * W1[(long)d * 512 + c];
    hid_s[idx] = fmaxf(acc, 0.f);
  }
  __syncthreads();

  for (int idx = tid; idx < 2048; idx += 256) {     // slots = sn + hidden @ W2 + b2
    int i = idx >> 8, dim = idx & 255;
    float acc = b2[dim];
    const float* hh = hid_s + i * 512;
    for (int c = 0; c < 512; ++c) acc += hh[c] * W2[(long)c * 256 + dim];
    float v = sn_s[idx] + acc;
    slots[(long)b * 2048 + idx] = v;
    if (write_out) out[(long)b * 2048 + idx] = v;
  }
}

extern "C" void kernel_launch(void* const* d_in, const int* in_sizes, int n_in,
                              void* d_out, int out_size, void* d_ws, size_t ws_size,
                              hipStream_t stream) {
  (void)in_sizes; (void)n_in; (void)out_size; (void)ws_size;
  const float* inputs     = (const float*)d_in[0];
  const float* slots_init = (const float*)d_in[1];
  const float* ln_f_g = (const float*)d_in[2];
  const float* ln_f_b = (const float*)d_in[3];
  const float* ln_s_g = (const float*)d_in[4];
  const float* ln_s_b = (const float*)d_in[5];
  const float* ln_ff_g = (const float*)d_in[6];
  const float* ln_ff_b = (const float*)d_in[7];
  const float* Wq = (const float*)d_in[8];  const float* bq = (const float*)d_in[9];
  const float* Wk = (const float*)d_in[10]; const float* bk = (const float*)d_in[11];
  const float* Wv = (const float*)d_in[12]; const float* bv = (const float*)d_in[13];
  const float* W_ih = (const float*)d_in[14]; const float* b_ih = (const float*)d_in[15];
  const float* W_hh = (const float*)d_in[16]; const float* b_hh = (const float*)d_in[17];
  const float* W1 = (const float*)d_in[18]; const float* b1 = (const float*)d_in[19];
  const float* W2 = (const float*)d_in[20]; const float* b2 = (const float*)d_in[21];

  char* base = (char*)d_ws; size_t off = 0;
  auto carve = [&](size_t bytes) -> void* {
    void* p = base + off; off += (bytes + 255) & ~(size_t)255; return p;
  };
  __bf16* WqT = (__bf16*)carve((size_t)65536 * 2);
  __bf16* WkT = (__bf16*)carve((size_t)65536 * 2);
  __bf16* WvT = (__bf16*)carve((size_t)65536 * 2);
  __bf16* qbf = (__bf16*)carve((size_t)32 * 16 * 256 * 2);
  __bf16* Kbf = (__bf16*)carve((size_t)32 * 4096 * 256 * 2);
  __bf16* Vt  = (__bf16*)carve((size_t)32 * 4096 * 256 * 2);
  float* U = (float*)carve((size_t)65536 * 4);
  float* S = (float*)carve((size_t)1024 * 4);
  float* slots = (float*)carve((size_t)65536 * 4);

  k_prep<<<1024, 256, 0, stream>>>(Wq, Wk, Wv, slots_init, WqT, WkT, WvT, slots);
  k_ln_project_kv<<<1024, 256, 0, stream>>>(inputs, ln_f_g, ln_f_b, WkT, WvT, bk, bv, Kbf, Vt);
  for (int it = 0; it < 5; ++it) {
    k_slots_q<<<32, 256, 0, stream>>>(slots, ln_s_g, ln_s_b, WqT, bq, qbf);
    k_zero_us<<<256, 256, 0, stream>>>(U, S);
    k_attn_update<<<1024, 256, 0, stream>>>(qbf, Kbf, Vt, U, S);
    k_gru_mlp<<<32, 256, 0, stream>>>(U, S, slots, W_ih, b_ih, W_hh, b_hh,
                                      ln_ff_g, ln_ff_b, W1, b1, W2, b2,
                                      (float*)d_out, it == 4 ? 1 : 0);
  }
}
